// Encoder_352187318911
// MI455X (gfx1250) — matrix operand order repaired and checked by reading, not compiled, not run on hardware
//
#include <hip/hip_runtime.h>

// ============================================================================
// GNN encoder for MI455X (gfx1250, wave32, WMMA).
//
// Math notes (exact simplifications):
//  * All four Linear/GCN biases feed directly into training-mode BatchNorm:
//    a constant per-column shift cancels in (h - mean) and leaves var
//    unchanged -> biases are provably no-ops and are skipped.
//  * BN is folded to a per-column affine (scale, shift) computed from batch
//    statistics; the affine + activation is fused into the *next* GEMM's
//    A-operand load (and into the final output kernel).
//  * GEMMs run on v_wmma_f32_16x16x32_bf16 (f32 accumulate); A is converted
//    fp32->bf16 on the fly, B (weights) are repacked once into the exact
//    per-lane WMMA B-operand layout so each lane does one contiguous 32B load.
//  * Scatter (GCN propagate) uses one wave per edge: edge index is forced
//    wave-uniform (readfirstlane) so src/dst/norm become SMEM scalar loads;
//    feature gather is vectorized and the next edge's source row is
//    prefetched (global_prefetch_b8).
// ============================================================================

typedef __attribute__((ext_vector_type(8)))  float  v8f;
typedef __attribute__((ext_vector_type(16))) __bf16 v16bf;

#define ACT_NONE 0
#define ACT_ELU  1
#define ACT_RELU 2

// ---------------------------------------------------------------------------
// Repack weight matrix W[K,N] (fp32 row-major) into bf16 WMMA-B tiles.
// Tile (kt,nt) covers K=kt*32..+31, N=nt*16..+15. Within a tile, lane l<16
// holds column n=nt*16+l, K rows kt*32+{0..7,16..23} (elements 0..15); lane l>=16
// holds column n=nt*16+(l-16), K rows kt*32+{8..15,24..31}. Stored so each lane's 16
// bf16 values are contiguous: tile*512 + lane*16 + i.
// ---------------------------------------------------------------------------
__global__ void pack_w_kernel(const float* __restrict__ W, __bf16* __restrict__ out,
                              int K, int N) {
  int tid = blockIdx.x * blockDim.x + threadIdx.x;
  if (tid >= K * N) return;
  int k = tid / N, n = tid - k * N;
  int kt = k >> 5, kin = k & 31;
  int nt = n >> 4, nin = n & 15;
  int lane = (((kin >> 3) & 1) << 4) | nin;
  int i = (kin & 7) | ((kin >> 4) << 3);
  size_t o = (size_t)(kt * (N >> 4) + nt) * 512 + (size_t)lane * 16 + i;
  out[o] = (__bf16)W[tid];
}

// ---------------------------------------------------------------------------
// WMMA GEMM: C[M,NTOT] = act(affine(A[M,KTOT])) @ Bpacked, fp32 out.
// One wave (32 threads) per 16-row M tile; the wave keeps NTOT/16 accumulator
// tiles so A is loaded from memory exactly once per row.
// A-operand layout (16x32 bf16): lane<16 -> row M=lane, K = {0..7, 16..23};
// lane>=16 -> row M=lane-16, K = {8..15, 24..31}.
// ---------------------------------------------------------------------------
template<int KTOT, int NTOT, int ACT>
__global__ __launch_bounds__(32)
void gemm_wmma_kernel(const float* __restrict__ A, const __bf16* __restrict__ Bp,
                      float* __restrict__ C,
                      const float* __restrict__ scale, const float* __restrict__ shift) {
  constexpr int NT = NTOT / 16;
  const int lane  = threadIdx.x;
  const int mtile = blockIdx.x;
  const int row   = mtile * 16 + (lane & 15);
  const int koff  = (lane >> 4) * 8;
  const float* __restrict__ arow = A + (size_t)row * KTOT;

  v8f zero = {};
  v8f acc[NT];
#pragma unroll
  for (int t = 0; t < NT; ++t) acc[t] = zero;

  for (int k0 = 0; k0 < KTOT; k0 += 32) {
    const float4 f0 = *reinterpret_cast<const float4*>(arow + k0 + koff);
    const float4 f1 = *reinterpret_cast<const float4*>(arow + k0 + koff + 4);
    const float4 f2 = *reinterpret_cast<const float4*>(arow + k0 + koff + 16);
    const float4 f3 = *reinterpret_cast<const float4*>(arow + k0 + koff + 20);
    float buf[16] = {f0.x, f0.y, f0.z, f0.w, f1.x, f1.y, f1.z, f1.w,
                     f2.x, f2.y, f2.z, f2.w, f3.x, f3.y, f3.z, f3.w};
    v16bf a;
#pragma unroll
    for (int i = 0; i < 16; ++i) {
      float v = buf[i];
      if (ACT != ACT_NONE) {
        int col = k0 + koff + ((i < 8) ? i : (8 + i));
        v = v * scale[col] + shift[col];
        if (ACT == ACT_ELU)  v = (v > 0.f) ? v : (__expf(v) - 1.f);
        if (ACT == ACT_RELU) v = fmaxf(v, 0.f);
      }
      a[i] = (__bf16)v;
    }
    const __bf16* btile = Bp + (size_t)((k0 >> 5) * NT) * 512 + (size_t)lane * 16;
#pragma unroll
    for (int t = 0; t < NT; ++t) {
      v16bf b = *reinterpret_cast<const v16bf*>(btile + (size_t)t * 512);
      acc[t] = __builtin_amdgcn_wmma_f32_16x16x32_bf16(
          /*neg_a=*/false, a, /*neg_b=*/false, b,
          /*c_mod=*/(short)0, acc[t], /*reuse_a=*/false, /*reuse_b=*/false);
    }
  }
  // C/D layout: VGPR j -> M = j + 8*(lane>=16), N = lane&15 (per 16-col tile).
  const int mrow = mtile * 16 + (lane >> 4) * 8;
  const int ncol = lane & 15;
#pragma unroll
  for (int t = 0; t < NT; ++t) {
    float* crow = C + (size_t)mrow * NTOT + t * 16 + ncol;
#pragma unroll
    for (int j = 0; j < 8; ++j) crow[(size_t)j * NTOT] = acc[t][j];
  }
}

// ---------------------------------------------------------------------------
// BatchNorm batch statistics: column-parallel sum / sum-of-squares reduction.
// Block = C threads (one column each), coalesced row reads, atomic combine.
// sums[0..C-1] = sum, sums[C..2C-1] = sumsq (must be pre-zeroed).
// Intermediates are L2-resident (51 MB max vs 192 MB L2), so the re-read is
// cheap; contended-atomic fusion into the GEMM epilogue is not worth it.
// ---------------------------------------------------------------------------
template<int C>
__global__ void bn_stats_kernel(const float* __restrict__ X, int M,
                                float* __restrict__ sums) {
  int c = threadIdx.x;
  float s = 0.f, s2 = 0.f;
  for (int r = blockIdx.x; r < M; r += gridDim.x) {
    float v = X[(size_t)r * C + c];
    s += v; s2 += v * v;
  }
  atomicAdd(&sums[c], s);
  atomicAdd(&sums[C + c], s2);
}

template<int C>
__global__ void bn_finalize_kernel(const float* __restrict__ sums,
                                   const float* __restrict__ gamma,
                                   const float* __restrict__ beta,
                                   float eps, float invM,
                                   float* __restrict__ scale, float* __restrict__ shift) {
  int c = threadIdx.x;
  float mean = sums[c] * invM;
  float var  = sums[C + c] * invM - mean * mean;   // biased var (matches jnp.var)
  float sc   = gamma[c] * rsqrtf(var + eps);
  scale[c] = sc;
  shift[c] = beta[c] - mean * sc;
}

// ---------------------------------------------------------------------------
// Graph plumbing: degrees (with self-loop), dinv = rsqrt(deg), edge norms.
// ---------------------------------------------------------------------------
__global__ void zero_f32_kernel(float* __restrict__ p, long long n) {
  long long i = (long long)blockIdx.x * blockDim.x + threadIdx.x;
  if (i < n) p[i] = 0.f;
}

__global__ void deg_init_kernel(float* __restrict__ deg, int n) {
  int i = blockIdx.x * blockDim.x + threadIdx.x;
  if (i < n) deg[i] = 1.0f;                         // self-loop contribution
}

__global__ void deg_count_kernel(const int* __restrict__ dst, int E,
                                 float* __restrict__ deg) {
  int e = blockIdx.x * blockDim.x + threadIdx.x;
  if (e < E) atomicAdd(&deg[dst[e]], 1.0f);
}

__global__ void deg_rsqrt_kernel(const float* __restrict__ deg,
                                 float* __restrict__ dinv, int n) {
  int i = blockIdx.x * blockDim.x + threadIdx.x;
  if (i < n) dinv[i] = rsqrtf(deg[i]);              // deg >= 1 always (self-loop)
}

__global__ void edge_norm_kernel(const int* __restrict__ src, const int* __restrict__ dst,
                                 const float* __restrict__ dinv,
                                 float* __restrict__ normw, int E) {
  int e = blockIdx.x * blockDim.x + threadIdx.x;
  if (e < E) normw[e] = dinv[src[e]] * dinv[dst[e]];
}

// ---------------------------------------------------------------------------
// D^-1/2 A D^-1/2 scatter-add (edges + self-loops). ONE WAVE PER EDGE:
// the edge index is forced wave-uniform via readfirstlane, so src/dst/normw
// resolve to SMEM scalar loads (1 per wave instead of 32 lane loads). The
// feature row gather is vectorized (float2 for C=64, float for C=32) and the
// next edge's source row is prefetched into L2 (global_prefetch_b8).
// ---------------------------------------------------------------------------
template<int C>
__global__ void scatter_kernel(const float* __restrict__ P,
                               const int* __restrict__ src, const int* __restrict__ dst,
                               const float* __restrict__ normw,
                               const float* __restrict__ dinv,
                               float* __restrict__ out, int E, int N) {
  constexpr int VEC = C / 32;                       // 2 for C=64, 1 for C=32
  const int lane = threadIdx.x & 31;
  long long tid = (long long)blockIdx.x * blockDim.x + threadIdx.x;
  int e = __builtin_amdgcn_readfirstlane((int)(tid >> 5));   // wave-uniform
  if (e >= E + N) return;

  // prefetch the source row of the edge this block's waves will see next round
  constexpr int WPB = 8;                            // 256 threads = 8 waves
  int epf = e + WPB;
  if (epf < E) {
    int spf = src[epf];                             // scalar load (uniform)
    __builtin_prefetch(P + (size_t)spf * C, 0, 1);
  }

  int s, d; float w;
  if (e < E) {
    s = src[e]; d = dst[e]; w = normw[e];           // SMEM scalar loads
  } else {
    s = d = e - E;                                  // self-loop
    float di = dinv[s];
    w = di * di;
  }

  if (VEC == 2) {
    const float2 v = *reinterpret_cast<const float2*>(P + (size_t)s * C + lane * 2);
    float* o = out + (size_t)d * C + lane * 2;
    atomicAdd(o + 0, v.x * w);
    atomicAdd(o + 1, v.y * w);
  } else {
    float v = P[(size_t)s * C + lane];
    atomicAdd(out + (size_t)d * C + lane, v * w);
  }
}

// Final layer: out = BN4(agg2) with identity activation (C = 32).
__global__ void affine32_kernel(const float* __restrict__ X,
                                const float* __restrict__ scale,
                                const float* __restrict__ shift,
                                float* __restrict__ out, long long total) {
  long long idx = (long long)blockIdx.x * blockDim.x + threadIdx.x;
  if (idx >= total) return;
  int c = (int)(idx & 31);
  out[idx] = X[idx] * scale[c] + shift[c];
}

// ---------------------------------------------------------------------------
// Workspace layout (bytes), ~91 MB total with buffer reuse.
// ---------------------------------------------------------------------------
static constexpr size_t O_WPK   = 0;              // packed bf16 weights (159,744 B)
static constexpr size_t O_STATS = 163840;         // 4 layers x 512 floats
static constexpr size_t O_DEG   = 172032;         // 100000 f32
static constexpr size_t O_DINV  = 572032;         // 100000 f32
static constexpr size_t O_NORMW = 972032;         // 3.2M f32
static constexpr size_t O_BIGA  = 13772032;       // 51.2 MB: H1, then {P1, AGG1}
static constexpr size_t O_BIGB  = 64972032;       // 25.6 MB: H2, then {P2, AGG2}
// packed-weight element offsets (bf16 elements)
static constexpr size_t W1P = 0;       // 512x128 -> 128 tiles
static constexpr size_t W2P = 65536;   // 128x64  -> 16 tiles
static constexpr size_t G1P = 73728;   // 64x64   -> 8 tiles
static constexpr size_t G2P = 77824;   // 64x32   -> 4 tiles

static inline int cdiv(long long a, long long b) { return (int)((a + b - 1) / b); }

extern "C" void kernel_launch(void* const* d_in, const int* in_sizes, int n_in,
                              void* d_out, int out_size, void* d_ws, size_t ws_size,
                              hipStream_t stream) {
  const float* x   = (const float*)d_in[0];
  const int*   ei  = (const int*)  d_in[1];
  const float* W1  = (const float*)d_in[2];
  // d_in[3] (b1), d_in[7] (b2), d_in[11] (bg1), d_in[15] (bg2): cancel under BN.
  const float* g1  = (const float*)d_in[4];
  const float* be1 = (const float*)d_in[5];
  const float* W2  = (const float*)d_in[6];
  const float* g2  = (const float*)d_in[8];
  const float* be2 = (const float*)d_in[9];
  const float* Wg1 = (const float*)d_in[10];
  const float* g3  = (const float*)d_in[12];
  const float* be3 = (const float*)d_in[13];
  const float* Wg2 = (const float*)d_in[14];
  const float* g4  = (const float*)d_in[16];
  const float* be4 = (const float*)d_in[17];

  const int Nn = in_sizes[0] / 512;   // 100000 nodes (multiple of 16)
  const int E  = in_sizes[1] / 2;     // 3,200,000 edges
  const int* src = ei;
  const int* dst = ei + E;

  char*   ws    = (char*)d_ws;
  __bf16* wpk   = (__bf16*)(ws + O_WPK);
  float*  stats = (float*) (ws + O_STATS);   // layer L base: stats + L*512
  float*  DEG   = (float*) (ws + O_DEG);
  float*  DINV  = (float*) (ws + O_DINV);
  float*  NORMW = (float*) (ws + O_NORMW);
  float*  H1    = (float*) (ws + O_BIGA);                       // 100000x128
  float*  P1    = (float*) (ws + O_BIGA);                       // reuse (H1 dead)
  float*  AGG1  = (float*) (ws + O_BIGA + (size_t)25600000);    // 100000x64
  float*  H2    = (float*) (ws + O_BIGB);                       // 100000x64
  float*  P2    = (float*) (ws + O_BIGB);                       // reuse (H2 dead)
  float*  AGG2  = (float*) (ws + O_BIGB + (size_t)12800000);    // 100000x32
  float*  out   = (float*)d_out;

  const int mt = Nn / 16;  // 6250 M-tiles
  const float invM = 1.0f / (float)Nn;
  const long long EN = (long long)E + Nn;  // edges + self-loops

  // ---- one-time-per-call weight repack (tiny) ----
  pack_w_kernel<<<cdiv(512 * 128, 256), 256, 0, stream>>>(W1,  wpk + W1P, 512, 128);
  pack_w_kernel<<<cdiv(128 *  64, 256), 256, 0, stream>>>(W2,  wpk + W2P, 128,  64);
  pack_w_kernel<<<cdiv( 64 *  64, 256), 256, 0, stream>>>(Wg1, wpk + G1P,  64,  64);
  pack_w_kernel<<<cdiv( 64 *  32, 256), 256, 0, stream>>>(Wg2, wpk + G2P,  64,  32);

  // ---- encoder L1: H1 = x @ W1 ; BN1 stats ----
  gemm_wmma_kernel<512, 128, ACT_NONE><<<mt, 32, 0, stream>>>(x, wpk + W1P, H1, nullptr, nullptr);
  float* st1 = stats;
  zero_f32_kernel<<<1, 256, 0, stream>>>(st1, 256);
  bn_stats_kernel<128><<<512, 128, 0, stream>>>(H1, Nn, st1);
  bn_finalize_kernel<128><<<1, 128, 0, stream>>>(st1, g1, be1, 1e-3f, invM, st1 + 256, st1 + 384);

  // ---- encoder L2: H2 = ELU(BN1(H1)) @ W2 ; BN2 stats ----
  gemm_wmma_kernel<128, 64, ACT_ELU><<<mt, 32, 0, stream>>>(H1, wpk + W2P, H2, st1 + 256, st1 + 384);
  float* st2 = stats + 512;
  zero_f32_kernel<<<1, 256, 0, stream>>>(st2, 256);
  bn_stats_kernel<64><<<512, 64, 0, stream>>>(H2, Nn, st2);
  bn_finalize_kernel<64><<<1, 64, 0, stream>>>(st2, g2, be2, 1e-3f, invM, st2 + 256, st2 + 384);

  // ---- graph norm precompute ----
  deg_init_kernel <<<cdiv(Nn, 256), 256, 0, stream>>>(DEG, Nn);
  deg_count_kernel<<<cdiv(E, 256), 256, 0, stream>>>(dst, E, DEG);
  deg_rsqrt_kernel<<<cdiv(Nn, 256), 256, 0, stream>>>(DEG, DINV, Nn);
  edge_norm_kernel<<<cdiv(E, 256), 256, 0, stream>>>(src, dst, DINV, NORMW, E);

  // ---- gc1: P1 = ELU(BN2(H2)) @ Wg1 ; propagate ; BN3 stats ----
  gemm_wmma_kernel<64, 64, ACT_ELU><<<mt, 32, 0, stream>>>(H2, wpk + G1P, P1, st2 + 256, st2 + 384);
  zero_f32_kernel<<<cdiv((long long)Nn * 64, 256), 256, 0, stream>>>(AGG1, (long long)Nn * 64);
  scatter_kernel<64><<<cdiv(EN, 8), 256, 0, stream>>>(P1, src, dst, NORMW, DINV, AGG1, E, Nn);
  float* st3 = stats + 1024;
  zero_f32_kernel<<<1, 256, 0, stream>>>(st3, 256);
  bn_stats_kernel<64><<<512, 64, 0, stream>>>(AGG1, Nn, st3);
  bn_finalize_kernel<64><<<1, 64, 0, stream>>>(st3, g3, be3, 1e-5f, invM, st3 + 256, st3 + 384);

  // ---- gc2: P2 = ReLU(BN3(AGG1)) @ Wg2 ; propagate ; BN4 ; output ----
  gemm_wmma_kernel<64, 32, ACT_RELU><<<mt, 32, 0, stream>>>(AGG1, wpk + G2P, P2, st3 + 256, st3 + 384);
  zero_f32_kernel<<<cdiv((long long)Nn * 32, 256), 256, 0, stream>>>(AGG2, (long long)Nn * 32);
  scatter_kernel<32><<<cdiv(EN, 8), 256, 0, stream>>>(P2, src, dst, NORMW, DINV, AGG2, E, Nn);
  float* st4 = stats + 1536;
  zero_f32_kernel<<<1, 256, 0, stream>>>(st4, 256);
  bn_stats_kernel<32><<<512, 32, 0, stream>>>(AGG2, Nn, st4);
  bn_finalize_kernel<32><<<1, 32, 0, stream>>>(st4, g4, be4, 1e-5f, invM, st4 + 256, st4 + 384);

  affine32_kernel<<<cdiv((long long)Nn * 32, 256), 256, 0, stream>>>(
      AGG2, st4 + 256, st4 + 384, out, (long long)Nn * 32);
}
